// ProductQuantizer_47880295416498
// MI455X (gfx1250) — matrix operand order repaired and checked by reading, not compiled, not run on hardware
//
#include <hip/hip_runtime.h>
#include <hip/hip_bf16.h>

// MI455X (gfx1250) product-quantizer pipeline, all GEMMs via v_wmma_f32_16x16x32_f16.
// Fragments are loaded with 2x ds_load_b128 each (A natural-major, B staged transposed).
// Workspace layout (floats): proj[32768*256] | Q[32768*256] | scal[2] (l2sum, covsum)

typedef _Float16 v16h __attribute__((ext_vector_type(16)));
typedef _Float16 v8h  __attribute__((ext_vector_type(8)));
typedef _Float16 h4   __attribute__((ext_vector_type(4)));
typedef float    v8f  __attribute__((ext_vector_type(8)));

#define N_ROWS 32768
#define D_DIM  1024
#define P_DIM  256
#define NSEC   4
#define NCODE  1024
#define SECD   64

__device__ __forceinline__ v8f wmma_f16(v16h a, v16h b, v8f c) {
  // (neg_a, A, neg_b, B, c_mod, C, reuse_a, reuse_b)
  return __builtin_amdgcn_wmma_f32_16x16x32_f16(false, a, false, b, (short)0, c, false, false);
}

__device__ __forceinline__ v8f vzero8() {
  v8f z;
#pragma unroll
  for (int i = 0; i < 8; ++i) z[i] = 0.0f;
  return z;
}

__device__ __forceinline__ h4 cvt4(float4 v) {
  h4 o;
  o[0] = (_Float16)v.x; o[1] = (_Float16)v.y;
  o[2] = (_Float16)v.z; o[3] = (_Float16)v.w;
  return o;
}

// A fragment (16x32 f16, MxK) from row-major LDS tile [row][k], stride in halves.
// lane: m = lane&15, half = lane>>4. elems 0..7: k=koff+8*half+0..7 ; elems 8..15: +16.
// stride*2 and koff*2 must be multiples of 16 bytes.
__device__ __forceinline__ v16h load_frag_a(const _Float16* lds, int stride, int row0,
                                            int koff, int lane) {
  const int m = row0 + (lane & 15);
  const int half = lane >> 4;
  const _Float16* base = lds + m * stride + koff + 8 * half;
  union { v16h v; v8h p[2]; } u;
  u.p[0] = *(const v8h*)(base);
  u.p[1] = *(const v8h*)(base + 16);
  return u.v;
}

// B fragment (32x16 f16, KxN) from TRANSPOSED LDS tile [n][k], stride in halves.
// lane: n = lane&15, half = lane>>4. elems 0..7: k=koff+8*half+0..7 ; elems 8..15: +16.
__device__ __forceinline__ v16h load_frag_bt(const _Float16* lds, int stride, int col0,
                                             int koff, int lane) {
  const int n = col0 + (lane & 15);
  const int half = lane >> 4;
  const _Float16* base = lds + n * stride + koff + 8 * half;
  union { v16h v; v8h p[2]; } u;
  u.p[0] = *(const v8h*)(base);
  u.p[1] = *(const v8h*)(base + 16);
  return u.v;
}

// ---------------------------------------------------------------------------
// K0: init — zero loss region & scalars, copy codebook out, cluster_counts=1
// ---------------------------------------------------------------------------
__global__ void k_init(float* __restrict__ loss, float* __restrict__ cbout,
                       const float* __restrict__ cbin, float* __restrict__ cnt,
                       float* __restrict__ scal) {
  size_t i = (size_t)blockIdx.x * blockDim.x + threadIdx.x;
  if (i < (size_t)N_ROWS * SECD) loss[i] = 0.0f;
  if (i < (size_t)NSEC * NCODE * SECD) cbout[i] = cbin[i];
  if (i < (size_t)NSEC * NCODE) cnt[i] = 1.0f;
  if (i < 2) scal[i] = 0.0f;
}

// ---------------------------------------------------------------------------
// K1: projected = (inp + bias) @ pca   [32768,1024]x[1024,256] -> [32768,256]
// grid (512,2), block 256; BM=64 BN=128 BK=32, wave grid 2x4, wave tile 32x32
// ---------------------------------------------------------------------------
__global__ __launch_bounds__(256) void k_project(const float* __restrict__ inp,
                                                 const float* __restrict__ pca,
                                                 const float* __restrict__ bias,
                                                 float* __restrict__ proj) {
  __shared__ _Float16 lA[64 * 40];
  __shared__ _Float16 lBt[128 * 40];
  const int tid = threadIdx.x, lane = tid & 31, w = tid >> 5;
  const int wm = w >> 2, wn = w & 3;
  const int m0 = blockIdx.x * 64, n0 = blockIdx.y * 128;
  const int half = lane >> 4, nl = lane & 15;
  v8f acc[2][2];
#pragma unroll
  for (int a = 0; a < 2; ++a)
#pragma unroll
    for (int b = 0; b < 2; ++b) acc[a][b] = vzero8();

  for (int kb = 0; kb < D_DIM; kb += 32) {
    // A: 64x32 f32 -> f16, float4 loads, h4 (b64) stores
#pragma unroll
    for (int i = tid; i < 512; i += 256) {
      int r = i >> 3, c = (i & 7) << 2;
      float4 v = *(const float4*)&inp[(size_t)(m0 + r) * D_DIM + kb + c];
      float4 bv = *(const float4*)&bias[kb + c];
      float4 t; t.x = v.x + bv.x; t.y = v.y + bv.y; t.z = v.z + bv.z; t.w = v.w + bv.w;
      *(h4*)&lA[r * 40 + c] = cvt4(t);
    }
    // B transposed: pca[k][n] -> lBt[n][k]; float4 in n, 4 scalar b16 stores
#pragma unroll
    for (int i = tid; i < 1024; i += 256) {
      int r = i >> 5, c = (i & 31) << 2;  // r = k, c = n
      float4 v = *(const float4*)&pca[(size_t)(kb + r) * P_DIM + n0 + c];
      lBt[(c + 0) * 40 + r] = (_Float16)v.x;
      lBt[(c + 1) * 40 + r] = (_Float16)v.y;
      lBt[(c + 2) * 40 + r] = (_Float16)v.z;
      lBt[(c + 3) * 40 + r] = (_Float16)v.w;
    }
    __syncthreads();
    v16h bf[2];
#pragma unroll
    for (int tn = 0; tn < 2; ++tn) bf[tn] = load_frag_bt(lBt, 40, wn * 32 + tn * 16, 0, lane);
#pragma unroll
    for (int tm = 0; tm < 2; ++tm) {
      v16h af = load_frag_a(lA, 40, wm * 32 + tm * 16, 0, lane);
#pragma unroll
      for (int tn = 0; tn < 2; ++tn) acc[tm][tn] = wmma_f16(af, bf[tn], acc[tm][tn]);
    }
    __syncthreads();
  }
#pragma unroll
  for (int tm = 0; tm < 2; ++tm)
#pragma unroll
    for (int tn = 0; tn < 2; ++tn)
#pragma unroll
      for (int r = 0; r < 8; ++r) {
        int mg = m0 + wm * 32 + tm * 16 + r + 8 * half;
        int ng = n0 + wn * 32 + tn * 16 + nl;
        proj[(size_t)mg * P_DIM + ng] = acc[tm][tn][r];
      }
}

// ---------------------------------------------------------------------------
// K2: fused  unprojected = proj @ Wt - bias ; per-row ||flat - unprojected||2
// ---------------------------------------------------------------------------
__global__ __launch_bounds__(256) void k_l2(const float* __restrict__ inp,
                                            const float* __restrict__ pca,
                                            const float* __restrict__ bias,
                                            const float* __restrict__ proj,
                                            float* __restrict__ scal) {
  __shared__ _Float16 lA[64 * 40];
  __shared__ _Float16 lBt[128 * 40];
  __shared__ float lRed[8][32][16];
  const int tid = threadIdx.x, lane = tid & 31, w = tid >> 5;
  const int wm = w >> 2, wn = w & 3;
  const int m0 = blockIdx.x * 64;
  const int half = lane >> 4, nl = lane & 15;
  float rs[16];
#pragma unroll
  for (int s = 0; s < 16; ++s) rs[s] = 0.0f;

  for (int nb = 0; nb < 8; ++nb) {
    const int n0 = nb * 128;
    v8f acc[2][2];
#pragma unroll
    for (int a = 0; a < 2; ++a)
#pragma unroll
      for (int b = 0; b < 2; ++b) acc[a][b] = vzero8();
    for (int kb = 0; kb < P_DIM; kb += 32) {
#pragma unroll
      for (int i = tid; i < 512; i += 256) {
        int r = i >> 3, c = (i & 7) << 2;
        float4 v = *(const float4*)&proj[(size_t)(m0 + r) * P_DIM + kb + c];
        *(h4*)&lA[r * 40 + c] = cvt4(v);
      }
      // Wt: pca[n][k] -> lBt[n][k]; contiguous both sides
#pragma unroll
      for (int i = tid; i < 1024; i += 256) {
        int c = i >> 3, k = (i & 7) << 2;  // c = n
        float4 v = *(const float4*)&pca[(size_t)(n0 + c) * P_DIM + kb + k];
        *(h4*)&lBt[c * 40 + k] = cvt4(v);
      }
      __syncthreads();
      v16h bf[2];
#pragma unroll
      for (int tn = 0; tn < 2; ++tn) bf[tn] = load_frag_bt(lBt, 40, wn * 32 + tn * 16, 0, lane);
#pragma unroll
      for (int tm = 0; tm < 2; ++tm) {
        v16h af = load_frag_a(lA, 40, wm * 32 + tm * 16, 0, lane);
#pragma unroll
        for (int tn = 0; tn < 2; ++tn) acc[tm][tn] = wmma_f16(af, bf[tn], acc[tm][tn]);
      }
      __syncthreads();
    }
#pragma unroll
    for (int tm = 0; tm < 2; ++tm)
#pragma unroll
      for (int tn = 0; tn < 2; ++tn)
#pragma unroll
        for (int r = 0; r < 8; ++r) {
          int mg = m0 + wm * 32 + tm * 16 + r + 8 * half;
          int ng = n0 + wn * 32 + tn * 16 + nl;
          float u = acc[tm][tn][r] - bias[ng];
          float d = inp[(size_t)mg * D_DIM + ng] - u;
          rs[tm * 8 + r] += d * d;
        }
  }
#pragma unroll
  for (int s = 0; s < 16; ++s) lRed[w][lane][s] = rs[s];
  __syncthreads();
  if (tid < 64) {
    int m = tid;
    int wr = m >> 5, tm = (m >> 4) & 1, hf = (m >> 3) & 1, r = m & 7;
    int slot = tm * 8 + r;
    float sum = 0.0f;
    for (int wc = 0; wc < 4; ++wc)
      for (int li = 0; li < 16; ++li) sum += lRed[wr * 4 + wc][hf * 16 + li][slot];
    atomicAdd(&scal[0], sqrtf(sum));
  }
}

// ---------------------------------------------------------------------------
// K3: cov = projT @ proj / N ; accumulate sum((cov - I)^2). grid (4,2).
// ---------------------------------------------------------------------------
__global__ __launch_bounds__(256) void k_cov(const float* __restrict__ proj,
                                             float* __restrict__ scal) {
  __shared__ _Float16 lA[64 * 40];
  __shared__ _Float16 lBt[128 * 40];
  __shared__ float red[256];
  const int tid = threadIdx.x, lane = tid & 31, w = tid >> 5;
  const int wm = w >> 2, wn = w & 3;
  const int m0 = blockIdx.x * 64, n0 = blockIdx.y * 128;
  const int half = lane >> 4, nl = lane & 15;
  v8f acc[2][2];
#pragma unroll
  for (int a = 0; a < 2; ++a)
#pragma unroll
    for (int b = 0; b < 2; ++b) acc[a][b] = vzero8();

  for (int kb = 0; kb < N_ROWS; kb += 32) {
    // A = projT: proj[k][m] -> lA[m][k]; float4 in m, 4 scalar b16 stores
#pragma unroll
    for (int i = tid; i < 512; i += 256) {
      int c = i >> 4, r = (i & 15) << 2;  // c = k, r = m
      float4 v = *(const float4*)&proj[(size_t)(kb + c) * P_DIM + m0 + r];
      lA[(r + 0) * 40 + c] = (_Float16)v.x;
      lA[(r + 1) * 40 + c] = (_Float16)v.y;
      lA[(r + 2) * 40 + c] = (_Float16)v.z;
      lA[(r + 3) * 40 + c] = (_Float16)v.w;
    }
    // B = proj: proj[k][n] -> lBt[n][k]; float4 in n, 4 scalar b16 stores
#pragma unroll
    for (int i = tid; i < 1024; i += 256) {
      int k = i >> 5, c = (i & 31) << 2;  // c = n
      float4 v = *(const float4*)&proj[(size_t)(kb + k) * P_DIM + n0 + c];
      lBt[(c + 0) * 40 + k] = (_Float16)v.x;
      lBt[(c + 1) * 40 + k] = (_Float16)v.y;
      lBt[(c + 2) * 40 + k] = (_Float16)v.z;
      lBt[(c + 3) * 40 + k] = (_Float16)v.w;
    }
    __syncthreads();
    v16h bf[2];
#pragma unroll
    for (int tn = 0; tn < 2; ++tn) bf[tn] = load_frag_bt(lBt, 40, wn * 32 + tn * 16, 0, lane);
#pragma unroll
    for (int tm = 0; tm < 2; ++tm) {
      v16h af = load_frag_a(lA, 40, wm * 32 + tm * 16, 0, lane);
#pragma unroll
      for (int tn = 0; tn < 2; ++tn) acc[tm][tn] = wmma_f16(af, bf[tn], acc[tm][tn]);
    }
    __syncthreads();
  }
  float p = 0.0f;
#pragma unroll
  for (int tm = 0; tm < 2; ++tm)
#pragma unroll
    for (int tn = 0; tn < 2; ++tn)
#pragma unroll
      for (int r = 0; r < 8; ++r) {
        int mg = m0 + wm * 32 + tm * 16 + r + 8 * half;
        int ng = n0 + wn * 32 + tn * 16 + nl;
        float c = acc[tm][tn][r] * (1.0f / (float)N_ROWS) - (mg == ng ? 1.0f : 0.0f);
        p += c * c;
      }
  red[tid] = p;
  __syncthreads();
  for (int s = 128; s > 0; s >>= 1) {
    if (tid < s) red[tid] += red[tid + s];
    __syncthreads();
  }
  if (tid == 0) atomicAdd(&scal[1], red[0]);
}

// ---------------------------------------------------------------------------
// K4: per-section VQ. dist = ||x||^2 - 2 x.c + ||c||^2 via WMMA, argmin over
// 1024 codes, gather q, write Q section + nn_idx + 0.25*(q-x)^2 into loss.
// grid (512, 4), block 256. 64 rows/block, code chunks of 128, K=64 (2 ksteps).
// ---------------------------------------------------------------------------
__global__ __launch_bounds__(256) void k_vq(const float* __restrict__ proj,
                                            const float* __restrict__ cbs,
                                            float* __restrict__ Q,
                                            float* __restrict__ loss,
                                            int* __restrict__ nnidx) {
  __shared__ _Float16 lA[64 * 72];
  __shared__ _Float16 lBt[128 * 72];
  __shared__ float xx[64];
  __shared__ float cc[128];
  __shared__ float bval[8][32][16];
  __shared__ int   bidxs[8][32][16];
  const int tid = threadIdx.x, lane = tid & 31, w = tid >> 5;
  const int wm = w >> 2, wn = w & 3;
  const int half = lane >> 4, nl = lane & 15;
  const int s = blockIdx.y;
  const int m0 = blockIdx.x * 64;

  // A: 64 rows x 64 dims of this section, contiguous both sides
#pragma unroll
  for (int i = tid; i < 1024; i += 256) {
    int r = i >> 4, c = (i & 15) << 2;
    float4 v = *(const float4*)&proj[(size_t)(m0 + r) * P_DIM + s * SECD + c];
    *(h4*)&lA[r * 72 + c] = cvt4(v);
  }
  __syncthreads();
  if (tid < 64) {
    float t = 0.0f;
    for (int c = 0; c < SECD; ++c) { float v = (float)lA[tid * 72 + c]; t += v * v; }
    xx[tid] = t;
  }
  v16h aF[2][2];
#pragma unroll
  for (int tm = 0; tm < 2; ++tm)
#pragma unroll
    for (int ks = 0; ks < 2; ++ks)
      aF[tm][ks] = load_frag_a(lA, 72, wm * 32 + tm * 16, ks * 32, lane);
  __syncthreads();

  float bv[16]; int bi[16];
#pragma unroll
  for (int q = 0; q < 16; ++q) { bv[q] = 3.4e38f; bi[q] = 0; }

  for (int cbk = 0; cbk < 8; ++cbk) {
    const int c0 = cbk * 128;
    // codebook chunk: cbs[code][k] -> lBt[n][k]; contiguous both sides
#pragma unroll
    for (int i = tid; i < 2048; i += 256) {
      int n = i >> 4, k = (i & 15) << 2;
      float4 v = *(const float4*)&cbs[((size_t)s * NCODE + c0 + n) * SECD + k];
      *(h4*)&lBt[n * 72 + k] = cvt4(v);
    }
    __syncthreads();
    if (tid < 128) {
      float t = 0.0f;
      for (int k = 0; k < SECD; ++k) { float v = (float)lBt[tid * 72 + k]; t += v * v; }
      cc[tid] = t;
    }
    __syncthreads();
    v8f acc[2][2];
#pragma unroll
    for (int a = 0; a < 2; ++a)
#pragma unroll
      for (int b = 0; b < 2; ++b) acc[a][b] = vzero8();
#pragma unroll
    for (int ks = 0; ks < 2; ++ks) {
      v16h bf[2];
#pragma unroll
      for (int tn = 0; tn < 2; ++tn)
        bf[tn] = load_frag_bt(lBt, 72, wn * 32 + tn * 16, ks * 32, lane);
#pragma unroll
      for (int tm = 0; tm < 2; ++tm)
#pragma unroll
        for (int tn = 0; tn < 2; ++tn)
          acc[tm][tn] = wmma_f16(aF[tm][ks], bf[tn], acc[tm][tn]);
    }
#pragma unroll
    for (int tn = 0; tn < 2; ++tn) {
      int nloc = wn * 32 + tn * 16 + nl;
      float ccv = cc[nloc];
      int gidx = c0 + nloc;
#pragma unroll
      for (int tm = 0; tm < 2; ++tm)
#pragma unroll
        for (int r = 0; r < 8; ++r) {
          int mloc = wm * 32 + tm * 16 + r + 8 * half;
          float dist = xx[mloc] - 2.0f * acc[tm][tn][r] + ccv;
          int slot = tm * 8 + r;
          if (dist < bv[slot]) { bv[slot] = dist; bi[slot] = gidx; }
        }
    }
    __syncthreads();
  }
#pragma unroll
  for (int q = 0; q < 16; ++q) { bval[w][lane][q] = bv[q]; bidxs[w][lane][q] = bi[q]; }
  __syncthreads();
  if (tid < 64) {
    int m = tid;
    int wr = m >> 5, tm = (m >> 4) & 1, hf = (m >> 3) & 1, r = m & 7;
    int slot = tm * 8 + r;
    float best = 3.4e38f; int bidx = 0x7fffffff;
    for (int wc = 0; wc < 4; ++wc)
      for (int li = 0; li < 16; ++li) {
        int ww = wr * 4 + wc, ll = hf * 16 + li;
        float v = bval[ww][ll][slot]; int ix = bidxs[ww][ll][slot];
        if (v < best || (v == best && ix < bidx)) { best = v; bidx = ix; }
      }
    int row = m0 + m;
    nnidx[(size_t)s * N_ROWS + row] = bidx;
    const float* qrow = &cbs[((size_t)s * NCODE + bidx) * SECD];
    for (int j = 0; j < SECD; ++j) {
      float qj = qrow[j];
      float xj = proj[(size_t)row * P_DIM + s * SECD + j];
      Q[(size_t)row * P_DIM + s * SECD + j] = qj;
      float d = qj - xj;
      atomicAdd(&loss[(size_t)row * SECD + j], 0.25f * d * d);
    }
  }
}

// ---------------------------------------------------------------------------
// K5: quantized_out = Q @ Wt - bias   [32768,256]x[256,1024] -> d_out
// ---------------------------------------------------------------------------
__global__ __launch_bounds__(256) void k_unproject(const float* __restrict__ Q,
                                                   const float* __restrict__ pca,
                                                   const float* __restrict__ bias,
                                                   float* __restrict__ out) {
  __shared__ _Float16 lA[64 * 40];
  __shared__ _Float16 lBt[128 * 40];
  const int tid = threadIdx.x, lane = tid & 31, w = tid >> 5;
  const int wm = w >> 2, wn = w & 3;
  const int m0 = blockIdx.x * 64, n0 = blockIdx.y * 128;
  const int half = lane >> 4, nl = lane & 15;
  v8f acc[2][2];
#pragma unroll
  for (int a = 0; a < 2; ++a)
#pragma unroll
    for (int b = 0; b < 2; ++b) acc[a][b] = vzero8();

  for (int kb = 0; kb < P_DIM; kb += 32) {
#pragma unroll
    for (int i = tid; i < 512; i += 256) {
      int r = i >> 3, c = (i & 7) << 2;
      float4 v = *(const float4*)&Q[(size_t)(m0 + r) * P_DIM + kb + c];
      *(h4*)&lA[r * 40 + c] = cvt4(v);
    }
#pragma unroll
    for (int i = tid; i < 1024; i += 256) {
      int c = i >> 3, k = (i & 7) << 2;  // c = n
      float4 v = *(const float4*)&pca[(size_t)(n0 + c) * P_DIM + kb + k];
      *(h4*)&lBt[c * 40 + k] = cvt4(v);
    }
    __syncthreads();
    v16h bf[2];
#pragma unroll
    for (int tn = 0; tn < 2; ++tn) bf[tn] = load_frag_bt(lBt, 40, wn * 32 + tn * 16, 0, lane);
#pragma unroll
    for (int tm = 0; tm < 2; ++tm) {
      v16h af = load_frag_a(lA, 40, wm * 32 + tm * 16, 0, lane);
#pragma unroll
      for (int tn = 0; tn < 2; ++tn) acc[tm][tn] = wmma_f16(af, bf[tn], acc[tm][tn]);
    }
    __syncthreads();
  }
#pragma unroll
  for (int tm = 0; tm < 2; ++tm)
#pragma unroll
    for (int tn = 0; tn < 2; ++tn)
#pragma unroll
      for (int r = 0; r < 8; ++r) {
        int mg = m0 + wm * 32 + tm * 16 + r + 8 * half;
        int ng = n0 + wn * 32 + tn * 16 + nl;
        out[(size_t)mg * D_DIM + ng] = acc[tm][tn][r] - bias[ng];
      }
}

// ---------------------------------------------------------------------------
// K6: loss += l2sum/N + covsum/P^2
// ---------------------------------------------------------------------------
__global__ void k_finalize(float* __restrict__ loss, const float* __restrict__ scal) {
  size_t i = (size_t)blockIdx.x * blockDim.x + threadIdx.x;
  float pca_loss = scal[0] * (1.0f / (float)N_ROWS) + scal[1] * (1.0f / (float)(P_DIM * P_DIM));
  if (i < (size_t)N_ROWS * SECD) loss[i] += pca_loss;
}

// ---------------------------------------------------------------------------
extern "C" void kernel_launch(void* const* d_in, const int* in_sizes, int n_in,
                              void* d_out, int out_size, void* d_ws, size_t ws_size,
                              hipStream_t stream) {
  const float* inp  = (const float*)d_in[0];  // [32,1024,1024]
  const float* pca  = (const float*)d_in[1];  // [1024,256]
  const float* bias = (const float*)d_in[2];  // [1,1024]
  const float* cbs  = (const float*)d_in[3];  // [4,1024,64]

  float* out     = (float*)d_out;
  float* o_quant = out;                                    // 33,554,432 f32
  float* o_loss  = out + (size_t)N_ROWS * D_DIM;           // 2,097,152 f32
  int*   o_idx   = (int*)(o_loss + (size_t)N_ROWS * SECD); // 131,072 i32
  float* o_cb    = (float*)(o_idx + (size_t)NSEC * N_ROWS);// 262,144 f32
  float* o_cnt   = o_cb + (size_t)NSEC * NCODE * SECD;     // 4,096 f32

  float* proj = (float*)d_ws;                      // [32768,256]
  float* Q    = proj + (size_t)N_ROWS * P_DIM;     // [32768,256]
  float* scal = Q + (size_t)N_ROWS * P_DIM;        // [2]: l2sum, covsum

  k_init<<<8192, 256, 0, stream>>>(o_loss, o_cb, cbs, o_cnt, scal);
  k_project<<<dim3(512, 2), 256, 0, stream>>>(inp, pca, bias, proj);
  k_l2<<<512, 256, 0, stream>>>(inp, pca, bias, proj, scal);
  k_cov<<<dim3(4, 2), 256, 0, stream>>>(proj, scal);
  k_vq<<<dim3(512, NSEC), 256, 0, stream>>>(proj, cbs, Q, o_loss, o_idx);
  k_unproject<<<dim3(512, 8), 256, 0, stream>>>(Q, pca, bias, o_quant);
  k_finalize<<<8192, 256, 0, stream>>>(o_loss, scal);
}
